// EncoderHead_23313082483042
// MI455X (gfx1250) — matrix operand order repaired and checked by reading, not compiled, not run on hardware
//
#include <hip/hip_runtime.h>

// EncoderHead: out = softmax((xWq)(xWk)^T * H^-0.5) @ (xWv)
// B=16, T=2048, C=H=64. Two-pass: f16 QKV into workspace, then wave-level
// flash attention with v_wmma_f32_16x16x32_f16 (fp32 accumulate).
//
// Workspace layout (f16): Q [B*T,64] | K [B*T,64] | Vt [B][64][T]  = 12 MB.

typedef __attribute__((ext_vector_type(16))) _Float16 v16h;
typedef __attribute__((ext_vector_type(8)))  _Float16 v8h;
typedef __attribute__((ext_vector_type(8)))  float    v8f;

#define WMMA_F16(a, b, c) \
  __builtin_amdgcn_wmma_f32_16x16x32_f16(false, (a), false, (b), (short)0, (c), false, false)

// A-matrix fragment (16x32, 16-bit) per ISA §7.12.2:
// lane m = lane&15; elements e<8 -> K = hi*8+e, e>=8 -> K = 16+hi*8+(e-8).
static __device__ __forceinline__ v16h load_fragA(const _Float16* base, int ld,
                                                  int row, int colbase, int hi) {
  const _Float16* p = base + row * ld + colbase + hi * 8;
  v8h lo = *(const v8h*)(p);
  v8h hh = *(const v8h*)(p + 16);
  v16h r;
#pragma unroll
  for (int e = 0; e < 8; ++e) { r[e] = lo[e]; r[e + 8] = hh[e]; }
  return r;
}

// B-matrix fragment (32x16, 16-bit), same K order as A:
// lane n = lane&15 holds column n; elements e<8 -> K = hi*8+e, e>=8 -> K = 16+hi*8+(e-8).
static __device__ __forceinline__ v16h load_fragB(const _Float16* base, int ld,
                                                  int row, int colbase, int hi) {
  const _Float16* p = base + row * ld + colbase + hi * 8;
  v8h lo = *(const v8h*)(p);
  v8h hh = *(const v8h*)(p + 16);
  v16h r;
#pragma unroll
  for (int e = 0; e < 8; ++e) { r[e] = lo[e]; r[e + 8] = hh[e]; }
  return r;
}

// A fragment from fp32 source (x), with cast to f16.
static __device__ __forceinline__ v16h load_fragA_f32(const float* base, int ld,
                                                      int row, int colbase, int hi) {
  const float* p = base + row * ld + colbase + hi * 8;
  v16h r;
#pragma unroll
  for (int e = 0; e < 8; ++e) { r[e] = (_Float16)p[e]; r[e + 8] = (_Float16)p[16 + e]; }
  return r;
}

static __device__ __forceinline__ float rmax16(float v) {
#pragma unroll
  for (int m = 1; m < 16; m <<= 1) v = fmaxf(v, __shfl_xor(v, m, 32));
  return v;
}
static __device__ __forceinline__ float rsum16(float v) {
#pragma unroll
  for (int m = 1; m < 16; m <<= 1) v += __shfl_xor(v, m, 32);
  return v;
}

// ---------------- Kernel 1: QKV projection ----------------
// 128 threads = 4 waves; each wave computes 16 rows x 64 cols of Q, K, V.
__global__ __launch_bounds__(128) void qkv_kernel(
    const float* __restrict__ x,
    const float* __restrict__ Wq, const float* __restrict__ Wk, const float* __restrict__ Wv,
    _Float16* __restrict__ Q, _Float16* __restrict__ K, _Float16* __restrict__ Vt) {
  __shared__ _Float16 Wt[3][64][72];  // transposed weights, padded stride
  const int tid = threadIdx.x;
  for (int idx = tid; idx < 3 * 4096; idx += 128) {
    int w = idx >> 12;
    int rem = idx & 4095;                        // rem = k*64 + n  (row-major W)
    const float* Wp = (w == 0) ? Wq : ((w == 1) ? Wk : Wv);
    Wt[w][rem & 63][rem >> 6] = (_Float16)Wp[rem];  // Wt[n][k] = W[k][n]
  }
  __syncthreads();

  const int lane = tid & 31, wave = tid >> 5;
  const int col = lane & 15, hi = lane >> 4;
  const int r0 = blockIdx.x * 64 + wave * 16;    // 512 blocks cover 32768 rows

  // A = x tile (16x64 fp32 -> f16), two K-chunks of 32
  v16h a0 = load_fragA_f32(x, 64, r0 + col, 0, hi);
  v16h a1 = load_fragA_f32(x, 64, r0 + col, 32, hi);

#pragma unroll
  for (int w = 0; w < 3; ++w) {
#pragma unroll
    for (int nt = 0; nt < 4; ++nt) {
      // B[k][n] = W[k][nt*16+n]  ->  lane n reads Wt row (nt*16+n), cols k
      v16h b0 = load_fragB(&Wt[w][0][0], 72, nt * 16 + col, 0, hi);
      v16h b1 = load_fragB(&Wt[w][0][0], 72, nt * 16 + col, 32, hi);
      v8f acc = {};
      acc = WMMA_F16(a0, b0, acc);
      acc = WMMA_F16(a1, b1, acc);
      const int oc = nt * 16 + col;
#pragma unroll
      for (int i = 0; i < 8; ++i) {              // D: element i -> row i + 8*hi
        int r = r0 + i + 8 * hi;
        float v = acc[i];
        if (w == 0) {
          Q[r * 64 + oc] = (_Float16)(v * 0.125f);     // fold H^-0.5 into Q
        } else if (w == 1) {
          K[r * 64 + oc] = (_Float16)v;
        } else {
          int b = r >> 11, t = r & 2047;
          Vt[((b << 6) + oc) * 2048 + t] = (_Float16)v; // V transposed [b][h][t]
        }
      }
    }
  }
}

// ---------------- Kernel 2: flash attention ----------------
// 256 threads = 8 waves; each wave owns 16 query rows, streams 64 keys/iter.
__global__ __launch_bounds__(256) void attn_kernel(
    const _Float16* __restrict__ Q, const _Float16* __restrict__ K,
    const _Float16* __restrict__ Vt, float* __restrict__ out) {
  __shared__ _Float16 Pbuf[8][16 * 72];          // per-wave 16x64 P tile, stride 72
  const int tid = threadIdx.x, lane = tid & 31, wave = tid >> 5;
  const int col = lane & 15, hi = lane >> 4;
  const int b = blockIdx.x >> 4, qt = blockIdx.x & 15;
  const int qrow = (b << 11) + qt * 128 + wave * 16;

  v16h aq0 = load_fragA(Q, 64, qrow + col, 0, hi);
  v16h aq1 = load_fragA(Q, 64, qrow + col, 32, hi);

  _Float16* P = &Pbuf[wave][0];
  const _Float16* Kb = K + ((size_t)b << 11) * 64;
  const _Float16* Vb = Vt + (size_t)b * 64 * 2048;

  float m8[8], l8[8];
  v8f o[4] = {{}, {}, {}, {}};
#pragma unroll
  for (int i = 0; i < 8; ++i) { m8[i] = -3.0e38f; l8[i] = 0.0f; }

  for (int kt = 0; kt < 2048; kt += 64) {
    // S = Q K^T for 16 q-rows x 64 keys: 4 D frags (16 keys each)
    v8f s[4];
#pragma unroll
    for (int nt = 0; nt < 4; ++nt) {
      // B[k][n] = Kmat[kt+nt*16+n][h=k]: lane n reads K row, contiguous h run
      v16h kb0 = load_fragB(Kb, 64, kt + nt * 16 + col, 0, hi);
      v16h kb1 = load_fragB(Kb, 64, kt + nt * 16 + col, 32, hi);
      v8f z = {};
      z = WMMA_F16(aq0, kb0, z);
      s[nt] = WMMA_F16(aq1, kb1, z);
    }

    // online softmax; row r = i + 8*hi spans 16 lanes of a half-wave
#pragma unroll
    for (int i = 0; i < 8; ++i) {
      float mx = fmaxf(fmaxf(s[0][i], s[1][i]), fmaxf(s[2][i], s[3][i]));
      mx = rmax16(mx);
      float mnew = fmaxf(m8[i], mx);
      float sc = __expf(m8[i] - mnew);
      float p0 = __expf(s[0][i] - mnew);
      float p1 = __expf(s[1][i] - mnew);
      float p2 = __expf(s[2][i] - mnew);
      float p3 = __expf(s[3][i] - mnew);
      float rs = rsum16((p0 + p1) + (p2 + p3));
      l8[i] = l8[i] * sc + rs;
      m8[i] = mnew;
      o[0][i] *= sc; o[1][i] *= sc; o[2][i] *= sc; o[3][i] *= sc;
      int pr = (i + 8 * hi) * 72 + col;          // D-layout -> LDS (row-major P)
      P[pr]      = (_Float16)p0;
      P[pr + 16] = (_Float16)p1;
      P[pr + 32] = (_Float16)p2;
      P[pr + 48] = (_Float16)p3;
    }
    asm volatile("s_wait_dscnt 0x0" ::: "memory");  // wave-private LDS RAW fence

    // reload P (16x64) as two A fragments; multiply by V (64 keys x 64 h)
    v16h pa0 = load_fragA(P, 72, col, 0, hi);
    v16h pa1 = load_fragA(P, 72, col, 32, hi);
#pragma unroll
    for (int nt = 0; nt < 4; ++nt) {
      // B[k][n] = V[kt+k][nt*16+n] = Vt[nt*16+n][kt+k]: contiguous key run
      v16h vb0 = load_fragB(Vb, 2048, nt * 16 + col, kt, hi);
      v16h vb1 = load_fragB(Vb, 2048, nt * 16 + col, kt + 32, hi);
      o[nt] = WMMA_F16(pa0, vb0, o[nt]);
      o[nt] = WMMA_F16(pa1, vb1, o[nt]);
    }
  }

#pragma unroll
  for (int i = 0; i < 8; ++i) {
    float inv = 1.0f / l8[i];
    int r = qrow + i + 8 * hi;
    float* op = out + (size_t)r * 64;
#pragma unroll
    for (int nt = 0; nt < 4; ++nt) op[nt * 16 + col] = o[nt][i] * inv;
  }
}

extern "C" void kernel_launch(void* const* d_in, const int* in_sizes, int n_in,
                              void* d_out, int out_size, void* d_ws, size_t ws_size,
                              hipStream_t stream) {
  const float* x  = (const float*)d_in[0];
  const float* Wq = (const float*)d_in[1];
  const float* Wk = (const float*)d_in[2];
  const float* Wv = (const float*)d_in[3];

  const size_t NROWS = (size_t)16 * 2048;        // B*T
  _Float16* Qws = (_Float16*)d_ws;
  _Float16* Kws = Qws + NROWS * 64;
  _Float16* Vt  = Kws + NROWS * 64;              // [B][64][2048]

  qkv_kernel<<<512, 128, 0, stream>>>(x, Wq, Wk, Wv, Qws, Kws, Vt);
  attn_kernel<<<256, 256, 0, stream>>>(Qws, Kws, Vt, (float*)d_out);
}
